// BSplineKANLayer_2886218023198
// MI455X (gfx1250) — matrix operand order repaired and checked by reading, not compiled, not run on hardware
//
#include <hip/hip_runtime.h>
#include <math.h>

typedef __attribute__((ext_vector_type(16))) __bf16 v16bf;
typedef __attribute__((ext_vector_type(8)))  __bf16 v8bf;
typedef __attribute__((ext_vector_type(4)))  __bf16 v4bf;
typedef __attribute__((ext_vector_type(8)))  float  v8f;
typedef __attribute__((ext_vector_type(4)))  float  v4f;
typedef __attribute__((ext_vector_type(4)))  int    v4i;

#define IN_DIM   1024
#define OUT_DIM  1024
#define BATCH    4096
#define NCOEF    8
#define NKNOT    12
#define KSPL     (IN_DIM * NCOEF)     // 8192
#define KTOT     (KSPL + IN_DIM)      // 9216
#define GSTRIDE  48                   // padded floats per grid-table row
#define EPSV     1e-8f

#define TILE_M   64
#define TILE_N   256
#define KC       32
#define NSTEPS   (KTOT / KC)          // 288
#define LPAD     40                   // LDS row stride in bf16 (32 + 8 pad)

// ---------------------------------------------------------------------------
// Async global->LDS copy (GLOBAL_LOAD_ASYNC_TO_LDS_B128, ASYNCcnt-tracked).
// Probe-confirmed param 0 type: 'v4i __device__*' (addrspace(1), int4 pointee).
// Falls back to a synchronous VGPR round-trip if the builtin is absent.
// ---------------------------------------------------------------------------
typedef __attribute__((address_space(1))) v4i* gbl_v4i_p;
typedef __attribute__((address_space(3))) v4i* lds_v4i_p;

#if __has_builtin(__builtin_amdgcn_global_load_async_to_lds_b128)
#define KAN_HAVE_ASYNC 1
__device__ __forceinline__ void async_cp16(void* lds, const void* g) {
  __builtin_amdgcn_global_load_async_to_lds_b128(
      (gbl_v4i_p)g, (lds_v4i_p)lds, 0, 0);
}
#else
#define KAN_HAVE_ASYNC 0
__device__ __forceinline__ void async_cp16(void* lds, const void* g) {
  *(v8bf*)lds = *(const v8bf*)g;
}
#endif

__device__ __forceinline__ void async_wait0() {
#if KAN_HAVE_ASYNC
#if __has_builtin(__builtin_amdgcn_s_wait_asynccnt)
  __builtin_amdgcn_s_wait_asynccnt(0);
#else
  asm volatile("s_wait_asynccnt 0x0" ::: "memory");
#endif
#endif
}

// ---------------------------------------------------------------------------
// Kernel 1: knots (softplus + cumsum) and Cox-de Boor denominator reciprocals.
// Row layout (48 f32): g[0..11], R1[12..22] (11), R2[23..32] (10), R3[33..41] (9)
// ---------------------------------------------------------------------------
__device__ __forceinline__ float softplusf(float v) {
  return fmaxf(v, 0.0f) + log1pf(expf(-fabsf(v)));
}

__global__ void kan_grid_prep(const float* __restrict__ gsl,
                              const float* __restrict__ gstart,
                              float* __restrict__ gtab) {
  int i = blockIdx.x * blockDim.x + threadIdx.x;
  if (i >= IN_DIM) return;
  float g[NKNOT];
  float a = gstart[i];
  g[0] = a;
#pragma unroll
  for (int j = 0; j < NKNOT - 1; ++j) {
    a += softplusf(gsl[i * (NKNOT - 1) + j]);
    g[j + 1] = a;
  }
  float* row = gtab + (size_t)i * GSTRIDE;
#pragma unroll
  for (int j = 0; j < NKNOT; ++j) row[j] = g[j];
#pragma unroll
  for (int t = 0; t < 11; ++t) row[12 + t] = 1.0f / (g[t + 1] - g[t] + EPSV);
#pragma unroll
  for (int t = 0; t < 10; ++t) row[23 + t] = 1.0f / (g[t + 2] - g[t] + EPSV);
#pragma unroll
  for (int t = 0; t < 9;  ++t) row[33 + t] = 1.0f / (g[t + 3] - g[t] + EPSV);
#pragma unroll
  for (int j = 42; j < GSTRIDE; ++j) row[j] = 0.0f;
}

// ---------------------------------------------------------------------------
// Kernel 2: combined bf16 weight matrix Ball[1024][9216]:
//   k <  8192 : coeffs[o][k]        (k = i*8 + c, matches basis ordering)
//   k >= 8192 : base_weight[o][k-8192]
// ---------------------------------------------------------------------------
__global__ void kan_bprep(const float* __restrict__ coeffs,
                          const float* __restrict__ basew,
                          __bf16* __restrict__ Ball) {
  int id = blockIdx.x * 256 + threadIdx.x;   // 1024 * (9216/4) threads
  int o  = id / (KTOT / 4);
  int k4 = (id % (KTOT / 4)) * 4;
  v4f v;
  if (k4 < KSPL) v = *(const v4f*)(coeffs + (size_t)o * KSPL + k4);
  else           v = *(const v4f*)(basew  + (size_t)o * IN_DIM + (k4 - KSPL));
  v4bf p;
#pragma unroll
  for (int j = 0; j < 4; ++j) p[j] = (__bf16)v[j];
  *(v4bf*)(Ball + (size_t)o * KTOT + k4) = p;
}

// ---------------------------------------------------------------------------
// Kernel 3: combined bf16 activation matrix Aall[4096][9216]:
//   k = i*8+c : order-3 B-spline basis_c(x[b][i])   (division-free via gtab)
//   k = 8192+i: silu(x[b][i])
// ---------------------------------------------------------------------------
__global__ void kan_aprep(const float* __restrict__ x,
                          const float* __restrict__ gtab,
                          __bf16* __restrict__ Aall) {
  int id = blockIdx.x * 256 + threadIdx.x;   // 4096*1024 threads
  int b = id >> 10;
  int i = id & (IN_DIM - 1);
  float xv = x[(size_t)b * IN_DIM + i];

  float gr[GSTRIDE];
  const v4f* grow = (const v4f*)(gtab + (size_t)i * GSTRIDE);
#pragma unroll
  for (int j = 0; j < GSTRIDE / 4; ++j) ((v4f*)gr)[j] = grow[j];
  const float* g  = gr;
  const float* R1 = gr + 12;
  const float* R2 = gr + 23;
  const float* R3 = gr + 33;

  float bb[11];
#pragma unroll
  for (int t = 0; t < 11; ++t)
    bb[t] = (xv >= g[t] && xv < g[t + 1]) ? 1.0f : 0.0f;
#pragma unroll
  for (int t = 0; t < 10; ++t)
    bb[t] = (xv - g[t]) * R1[t] * bb[t] + (g[t + 2] - xv) * R1[t + 1] * bb[t + 1];
#pragma unroll
  for (int t = 0; t < 9; ++t)
    bb[t] = (xv - g[t]) * R2[t] * bb[t] + (g[t + 3] - xv) * R2[t + 1] * bb[t + 1];
#pragma unroll
  for (int t = 0; t < 8; ++t)
    bb[t] = (xv - g[t]) * R3[t] * bb[t] + (g[t + 4] - xv) * R3[t + 1] * bb[t + 1];

  v8bf p;
#pragma unroll
  for (int e = 0; e < 8; ++e) p[e] = (__bf16)bb[e];
  *(v8bf*)(Aall + (size_t)b * KTOT + (size_t)i * NCOEF) = p;

  float s = xv / (1.0f + expf(-xv));
  Aall[(size_t)b * KTOT + KSPL + i] = (__bf16)s;
}

// ---------------------------------------------------------------------------
// Kernel 4: WMMA GEMM  out[4096][1024] = A(4096x9216,bf16) * B(1024x9216,bf16)^T
//           + res_scale * x.
// Block tile 64x256, 8 waves (2x4), wave tile 32x64 (2x4 WMMA accumulators).
// Double-buffered LDS fed by async global->LDS copies (ASYNCcnt), one barrier
// per K-step; async DMA of step s+1 overlaps ds_load+WMMA of step s.
// ---------------------------------------------------------------------------
__global__ __launch_bounds__(256)
void kan_gemm(const __bf16* __restrict__ A,
              const __bf16* __restrict__ Bm,
              const float* __restrict__ x,
              const float* __restrict__ res_scale,
              float* __restrict__ out) {
  __shared__ __bf16 sA[2][TILE_M * LPAD];   // 2 x 5120 B
  __shared__ __bf16 sB[2][TILE_N * LPAD];   // 2 x 20480 B

  const int t    = threadIdx.x;
  const int lane = t & 31;
  const int wave = t >> 5;      // 0..7
  const int wm   = wave >> 2;   // 0..1  (M direction)
  const int wn   = wave & 3;    // 0..3  (N direction)
  const int m0   = blockIdx.x * TILE_M;
  const int n0   = blockIdx.y * TILE_N;
  const int h    = lane >> 4;   // lane half
  const int ml   = lane & 15;

  // staging: thread t copies 16B of A row t/4 and 64B of B row t per step
  const __bf16* aG = A  + (size_t)(m0 + (t >> 2)) * KTOT + (t & 3) * 8;
  const __bf16* bG = Bm + (size_t)(n0 + t) * KTOT;
  const int sAoff = (t >> 2) * LPAD + (t & 3) * 8;
  const int sBoff = t * LPAD;

  v8f acc[2][4] = {};

  // prologue: stage step 0 into buffer 0
  {
    async_cp16(&sA[0][sAoff], aG);
#pragma unroll
    for (int j = 0; j < 4; ++j)
      async_cp16(&sB[0][sBoff + j * 8], bG + j * 8);
    async_wait0();
    __syncthreads();
  }

  for (int s = 0; s < NSTEPS; ++s) {
    const int cur = s & 1;
    // prefetch next K-slice into the other buffer (overlaps with WMMA below)
    if (s + 1 < NSTEPS) {
      const int k1 = (s + 1) * KC;
      async_cp16(&sA[cur ^ 1][sAoff], aG + k1);
#pragma unroll
      for (int j = 0; j < 4; ++j)
        async_cp16(&sB[cur ^ 1][sBoff + j * 8], bG + k1 + j * 8);
    }

    const __bf16* sAc = sA[cur];
    const __bf16* sBc = sB[cur];

    // A fragment: lane (M=ml) takes contiguous K run 16h..16h+15
    v16bf aF[2];
#pragma unroll
    for (int mi = 0; mi < 2; ++mi) {
      const __bf16* p = sAc + (wm * 32 + mi * 16 + ml) * LPAD + h * 16;
      v8bf lo = *(const v8bf*)p;
      v8bf hi = *(const v8bf*)(p + 8);
      aF[mi] = __builtin_shufflevector(lo, hi, 0,1,2,3,4,5,6,7,8,9,10,11,12,13,14,15);
    }
    // B fragment: lane = column N=ml, contiguous K run 16h..16h+15
    v16bf bF[4];
#pragma unroll
    for (int ni = 0; ni < 4; ++ni) {
      const __bf16* p = sBc + (wn * 64 + ni * 16 + ml) * LPAD + h * 16;
      v8bf lo = *(const v8bf*)p;
      v8bf hi = *(const v8bf*)(p + 8);
      bF[ni] = __builtin_shufflevector(lo, hi, 0,1,2,3,4,5,6,7,8,9,10,11,12,13,14,15);
    }

#pragma unroll
    for (int mi = 0; mi < 2; ++mi)
#pragma unroll
      for (int ni = 0; ni < 4; ++ni)
        acc[mi][ni] = __builtin_amdgcn_wmma_f32_16x16x32_bf16(
            false, aF[mi], false, bF[ni], (short)0, acc[mi][ni], false, false);

    async_wait0();     // this wave's DMA into the next buffer is complete
    __syncthreads();   // everyone done reading cur + done filling next
  }

  // Epilogue: D layout -> VGPR r: lanes 0-15 M=r, lanes 16-31 M=8+r; N=ml.
  // Batch the residual x loads so they issue as one clause.
  const float rs = res_scale[0];
#pragma unroll
  for (int mi = 0; mi < 2; ++mi) {
    const int rbase = m0 + wm * 32 + mi * 16 + h * 8;
#pragma unroll
    for (int ni = 0; ni < 4; ++ni) {
      const int col = n0 + wn * 64 + ni * 16 + ml;
      const float* xp = x   + (size_t)rbase * OUT_DIM + col;
      float*       op = out + (size_t)rbase * OUT_DIM + col;
      float xv[8];
#pragma unroll
      for (int r = 0; r < 8; ++r) xv[r] = xp[(size_t)r * OUT_DIM];
#pragma unroll
      for (int r = 0; r < 8; ++r) op[(size_t)r * OUT_DIM] = acc[mi][ni][r] + rs * xv[r];
    }
  }
}

// ---------------------------------------------------------------------------
// Launcher. Workspace: [gtab 192KB][Ball 18.9MB][Aall 75.5MB] ~= 90.2 MiB
// ---------------------------------------------------------------------------
extern "C" void kernel_launch(void* const* d_in, const int* in_sizes, int n_in,
                              void* d_out, int out_size, void* d_ws, size_t ws_size,
                              hipStream_t stream) {
  (void)in_sizes; (void)n_in; (void)out_size; (void)ws_size;
  const float* x      = (const float*)d_in[0];
  const float* coeffs = (const float*)d_in[1];
  const float* basew  = (const float*)d_in[2];
  const float* gsl    = (const float*)d_in[3];
  const float* gstart = (const float*)d_in[4];
  const float* res    = (const float*)d_in[5];
  float* out = (float*)d_out;

  char* ws = (char*)d_ws;
  const size_t gtab_bytes = (size_t)IN_DIM * GSTRIDE * sizeof(float);      // 196608
  const size_t ball_bytes = (size_t)OUT_DIM * KTOT * sizeof(__bf16);       // 18874368
  float*  gtab = (float*)ws;
  __bf16* Ball = (__bf16*)(ws + gtab_bytes);
  __bf16* Aall = (__bf16*)(ws + gtab_bytes + ball_bytes);

  kan_grid_prep<<<(IN_DIM + 255) / 256, 256, 0, stream>>>(gsl, gstart, gtab);
  kan_bprep<<<(OUT_DIM * (KTOT / 4)) / 256, 256, 0, stream>>>(coeffs, basew, Ball);
  kan_aprep<<<(BATCH * IN_DIM) / 256, 256, 0, stream>>>(x, gtab, Aall);

  dim3 grid(BATCH / TILE_M, OUT_DIM / TILE_N);
  kan_gemm<<<grid, 256, 0, stream>>>(Aall, Ball, x, res, out);
}
